// TinyModel_81149112091091
// MI455X (gfx1250) — matrix operand order repaired and checked by reading, not compiled, not run on hardware
//
#include <hip/hip_runtime.h>
#include <cstdint>
#include <cstddef>

#define VOCAB  2048
#define EMBED  512
#define HIDDEN 1024
#define GATES  4096      // 4*HIDDEN
#define NB     64        // batch
#define NTIME  512       // timesteps
#define MROWS  32768     // NB*NTIME

typedef __bf16    bf16x16 __attribute__((ext_vector_type(16)));
typedef float     f32x8   __attribute__((ext_vector_type(8)));
typedef uint32_t  u32x4   __attribute__((ext_vector_type(4)));

union TileAB { bf16x16 v; u32x4 q[2]; };

__device__ __forceinline__ uint16_t f2bf(float f) {
  union { float f; uint32_t u; } c; c.f = f;
  uint32_t u = c.u + 0x7fffu + ((c.u >> 16) & 1u);   // round-to-nearest-even
  return (uint16_t)(u >> 16);
}
__device__ __forceinline__ float bf2f(uint16_t h) {
  union { uint32_t u; float f; } c; c.u = ((uint32_t)h) << 16;
  return c.f;
}
__device__ __forceinline__ float sigmoidf_(float x) {
  return 1.0f / (1.0f + __expf(-x));
}

// A tile 16x32 bf16: lane<16 -> row=lane, K k0..k0+15;
// lane>=16 -> row=lane-16, K k0+16..k0+31. Two b128 loads.
__device__ __forceinline__ TileAB loadA(const uint16_t* rowptr, int k0, int lane) {
  const int kb = k0 + ((lane >> 4) << 4);
  const u32x4* p = (const u32x4*)(rowptr + kb);
  TileAB t; t.q[0] = p[0]; t.q[1] = p[1]; return t;
}
// B tile 32x16 bf16 from row-major W[N][K] (i.e. B = W^T):
// lane<16 -> col=lane, K k0..k0+15 ; lane>=16 -> col=lane-16, K k0+16..k0+31.
// 32 contiguous bytes per lane.
__device__ __forceinline__ TileAB loadB(const uint16_t* W, int ldk, int n0, int k0, int lane) {
  const int n  = n0 + (lane & 15);
  const int kb = k0 + ((lane >> 4) << 4);
  const u32x4* p = (const u32x4*)(W + (size_t)n * ldk + kb);
  TileAB t; t.q[0] = p[0]; t.q[1] = p[1]; return t;
}

// ---------------- prep kernels ----------------
__global__ void cvt_bf16_kernel(const float* __restrict__ src, uint16_t* __restrict__ dst, int n) {
  int i = blockIdx.x * blockDim.x + threadIdx.x;
  int stride = gridDim.x * blockDim.x;
  for (; i < n; i += stride) dst[i] = f2bf(src[i]);
}
__global__ void bias_sum_kernel(const float* __restrict__ a, const float* __restrict__ b,
                                float* __restrict__ out) {
  int i = blockIdx.x * blockDim.x + threadIdx.x;
  if (i < GATES) out[i] = a[i] + b[i];
}

// ---------------- K1: x_proj = gather(emb)[MROWS,512] @ W_ih^T + (b_ih+b_hh), bf16 out [T][B][4H]
__global__ __launch_bounds__(256) void xproj_gemm_kernel(
    const int* __restrict__ xids, const uint16_t* __restrict__ emb,
    const uint16_t* __restrict__ Wih, const float* __restrict__ bsum,
    uint16_t* __restrict__ xproj) {
  const int lane = threadIdx.x & 31, w = threadIdx.x >> 5;
  const int lm = lane & 15, lh = lane >> 4;
  const int gw = blockIdx.x * 8 + w;
  const int NT = GATES / 16;             // 256 N-tiles
  const int nt = gw % NT;
  const int mb = gw / NT;                // 64-row M block, 512 blocks
  const int n0 = nt * 16;

  const uint16_t* arow[4];
#pragma unroll
  for (int mi = 0; mi < 4; ++mi) {
    int r = mb * 64 + mi * 16 + lm;      // r = t*64 + b  (x_proj row order [T][B])
    int t = r >> 6, b = r & 63;
    arow[mi] = emb + (size_t)xids[b * NTIME + t] * EMBED;
  }
  f32x8 acc[4] = { {0,0,0,0,0,0,0,0}, {0,0,0,0,0,0,0,0},
                   {0,0,0,0,0,0,0,0}, {0,0,0,0,0,0,0,0} };
  for (int k0 = 0; k0 < EMBED; k0 += 32) {
    TileAB bt = loadB(Wih, EMBED, n0, k0, lane);
#pragma unroll
    for (int mi = 0; mi < 4; ++mi) {
      TileAB at = loadA(arow[mi], k0, lane);
      acc[mi] = __builtin_amdgcn_wmma_f32_16x16x32_bf16(
          false, at.v, false, bt.v, (short)0, acc[mi], false, false);
    }
  }
  const int col = n0 + lm;
  const float bias = bsum[col];
#pragma unroll
  for (int mi = 0; mi < 4; ++mi) {
    int r0 = mb * 64 + mi * 16 + lh * 8;   // C layout: row = 8*(lane>=16)+k, col = lane%16
#pragma unroll
    for (int k = 0; k < 8; ++k)
      xproj[(size_t)(r0 + k) * GATES + col] = f2bf(acc[mi][k] + bias);
  }
}

// ---------------- K2: persistent LSTM recurrence ----------------
// 64 WGs (one per 16 hidden units), 4 waves each (one per 16-batch M tile).
// W_hh slice (64 rows x 1024) staged in LDS once; h exchanged via global + grid barrier.
__global__ __launch_bounds__(128) void lstm_kernel(
    const uint16_t* __restrict__ xproj, const uint16_t* __restrict__ Whh,
    uint16_t* __restrict__ hbuf, uint16_t* __restrict__ hs,
    unsigned* __restrict__ counter) {
  extern __shared__ uint16_t smem[];      // 64*1024 bf16 = 128 KB
  const int wg = blockIdx.x;              // hidden cols [wg*16, wg*16+16)
  const int lane = threadIdx.x & 31, w = threadIdx.x >> 5;
  const int lm = lane & 15, lh = lane >> 4;

  // Stage W_hh rows {g*1024 + wg*16 + j : g<4, j<16} -> LDS rows g*16+j.
  for (int i = threadIdx.x; i < 64 * (HIDDEN / 8); i += blockDim.x) {
    int lr = i >> 7;                      // 128 u32x4 per row
    int kk = (i & 127) * 8;
    int grow = (lr >> 4) * HIDDEN + wg * 16 + (lr & 15);
    ((u32x4*)smem)[i] = *(const u32x4*)(Whh + (size_t)grow * HIDDEN + kk);
  }
  __syncthreads();

  const int colh = wg * 16 + lm;          // this lane's hidden-unit column
  const int rowb = w * 16 + lh * 8;       // this lane's batch-row base
  f32x8 cst = {0,0,0,0,0,0,0,0};          // cell state, in registers

  for (int t = 0; t < NTIME; ++t) {
    const uint16_t* hin = hbuf + (size_t)(t & 1) * NB * HIDDEN;
    uint16_t* hout = hbuf + (size_t)((t + 1) & 1) * NB * HIDDEN;
    const uint16_t* arow = hin + (size_t)(w * 16 + lm) * HIDDEN;

    f32x8 acc[4] = { {0,0,0,0,0,0,0,0}, {0,0,0,0,0,0,0,0},
                     {0,0,0,0,0,0,0,0}, {0,0,0,0,0,0,0,0} };
    for (int k0 = 0; k0 < HIDDEN; k0 += 32) {
      TileAB at = loadA(arow, k0, lane);
#pragma unroll
      for (int g = 0; g < 4; ++g) {
        TileAB bt;                         // B from LDS: rows g*16+lm, 32B contiguous
        const u32x4* p = (const u32x4*)(smem + (size_t)(g * 16 + lm) * HIDDEN
                                        + k0 + (lh << 4));
        bt.q[0] = p[0]; bt.q[1] = p[1];
        acc[g] = __builtin_amdgcn_wmma_f32_16x16x32_bf16(
            false, at.v, false, bt.v, (short)0, acc[g], false, false);
      }
    }
    // pointwise cell update (gate order i,f,g,o)
    const uint16_t* xp = xproj + (size_t)t * NB * GATES;
#pragma unroll
    for (int k = 0; k < 8; ++k) {
      int b = rowb + k;
      float gi = acc[0][k] + bf2f(xp[(size_t)b * GATES + 0 * HIDDEN + colh]);
      float gf = acc[1][k] + bf2f(xp[(size_t)b * GATES + 1 * HIDDEN + colh]);
      float gg = acc[2][k] + bf2f(xp[(size_t)b * GATES + 2 * HIDDEN + colh]);
      float go = acc[3][k] + bf2f(xp[(size_t)b * GATES + 3 * HIDDEN + colh]);
      float iv = sigmoidf_(gi), fv = sigmoidf_(gf);
      float gv = tanhf(gg),     ov = sigmoidf_(go);
      float cv = fv * cst[k] + iv * gv;
      cst[k] = cv;
      uint16_t hb = f2bf(ov * tanhf(cv));
      hout[(size_t)b * HIDDEN + colh] = hb;
      hs[((size_t)b * NTIME + t) * HIDDEN + colh] = hb;   // hs row = b*T + t
    }
    // device-wide barrier (release stores, arrive, spin, acquire)
    __builtin_amdgcn_fence(__ATOMIC_RELEASE, "agent");
    __syncthreads();
    if (threadIdx.x == 0) {
      __hip_atomic_fetch_add(counter, 1u, __ATOMIC_RELAXED, __HIP_MEMORY_SCOPE_AGENT);
      unsigned target = 64u * (unsigned)(t + 1);
      while (__hip_atomic_load(counter, __ATOMIC_RELAXED, __HIP_MEMORY_SCOPE_AGENT) < target)
        __builtin_amdgcn_s_sleep(1);
    }
    __syncthreads();
    __builtin_amdgcn_fence(__ATOMIC_ACQUIRE, "agent");
  }
}

// ---------------- K3: logits = hs[MROWS,1024] @ fc_W^T + fc_b, f32 out ----------------
__global__ __launch_bounds__(256) void logits_gemm_kernel(
    const uint16_t* __restrict__ hs, const uint16_t* __restrict__ fcW,
    const float* __restrict__ fcb, float* __restrict__ logits) {
  const int lane = threadIdx.x & 31, w = threadIdx.x >> 5;
  const int lm = lane & 15, lh = lane >> 4;
  const int gw = blockIdx.x * 8 + w;
  const int NT = VOCAB / 16;              // 128 N-tiles
  const int nt = gw % NT;
  const int mb = gw / NT;                 // 512 blocks of 64 rows
  const int n0 = nt * 16;

  const uint16_t* arow[4];
#pragma unroll
  for (int mi = 0; mi < 4; ++mi) {
    int r = mb * 64 + mi * 16 + lm;
    arow[mi] = hs + (size_t)r * HIDDEN;
  }
  f32x8 acc[4] = { {0,0,0,0,0,0,0,0}, {0,0,0,0,0,0,0,0},
                   {0,0,0,0,0,0,0,0}, {0,0,0,0,0,0,0,0} };
  for (int k0 = 0; k0 < HIDDEN; k0 += 32) {
    TileAB bt = loadB(fcW, HIDDEN, n0, k0, lane);
#pragma unroll
    for (int mi = 0; mi < 4; ++mi) {
      TileAB at = loadA(arow[mi], k0, lane);
      acc[mi] = __builtin_amdgcn_wmma_f32_16x16x32_bf16(
          false, at.v, false, bt.v, (short)0, acc[mi], false, false);
    }
  }
  const int col = n0 + lm;
  const float bias = fcb[col];
#pragma unroll
  for (int mi = 0; mi < 4; ++mi) {
    int r0 = mb * 64 + mi * 16 + lh * 8;
#pragma unroll
    for (int k = 0; k < 8; ++k)
      logits[(size_t)(r0 + k) * VOCAB + col] = acc[mi][k] + bias;
  }
}

extern "C" void kernel_launch(void* const* d_in, const int* in_sizes, int n_in,
                              void* d_out, int out_size, void* d_ws, size_t ws_size,
                              hipStream_t stream) {
  const int*   x    = (const int*)  d_in[0];
  const float* emb  = (const float*)d_in[1];
  const float* Wih  = (const float*)d_in[2];
  const float* Whh  = (const float*)d_in[3];
  const float* bih  = (const float*)d_in[4];
  const float* bhh  = (const float*)d_in[5];
  const float* fcW  = (const float*)d_in[6];
  const float* fcb  = (const float*)d_in[7];
  float* logits = (float*)d_out;

  char* ws = (char*)d_ws;
  unsigned* counter = (unsigned*)ws;                                   // 256 B slot
  uint16_t* emb_bf  = (uint16_t*)(ws + 256);
  uint16_t* Wih_bf  = emb_bf + (size_t)VOCAB * EMBED;
  uint16_t* Whh_bf  = Wih_bf + (size_t)GATES * EMBED;
  uint16_t* fcW_bf  = Whh_bf + (size_t)GATES * HIDDEN;
  float*    bsum    = (float*)(fcW_bf + (size_t)VOCAB * HIDDEN);
  uint16_t* hbuf    = (uint16_t*)(bsum + GATES);                       // 2 x [64][1024]
  uint16_t* hs      = hbuf + 2 * (size_t)NB * HIDDEN;                  // [MROWS][1024]
  uint16_t* xproj   = hs + (size_t)MROWS * HIDDEN;                     // [T][B][4H]

  hipMemsetAsync(counter, 0, 256, stream);
  hipMemsetAsync(hbuf, 0, 2 * (size_t)NB * HIDDEN * sizeof(uint16_t), stream);

  cvt_bf16_kernel<<<1024, 256, 0, stream>>>(emb, emb_bf, VOCAB * EMBED);
  cvt_bf16_kernel<<<1024, 256, 0, stream>>>(Wih, Wih_bf, GATES * EMBED);
  cvt_bf16_kernel<<<1024, 256, 0, stream>>>(Whh, Whh_bf, GATES * HIDDEN);
  cvt_bf16_kernel<<<1024, 256, 0, stream>>>(fcW, fcW_bf, VOCAB * HIDDEN);
  bias_sum_kernel<<<GATES / 256, 256, 0, stream>>>(bih, bhh, bsum);

  // 512 M-blocks * 256 N-tiles = 131072 waves / 8 per block
  xproj_gemm_kernel<<<16384, 256, 0, stream>>>(x, emb_bf, Wih_bf, bsum, xproj);

  // persistent recurrence: 64 WGs, 128 threads, 128 KB dynamic LDS
  lstm_kernel<<<64, 128, 64 * HIDDEN * sizeof(uint16_t), stream>>>(
      xproj, Whh_bf, hbuf, hs, counter);

  // 512 M-blocks * 128 N-tiles = 65536 waves / 8 per block
  logits_gemm_kernel<<<8192, 256, 0, stream>>>(hs, fcW_bf, fcb, logits);
}
